// HeteroModel_70248485094040
// MI455X (gfx1250) — matrix operand order repaired and checked by reading, not compiled, not run on hardware
//
#include <hip/hip_runtime.h>
#include <hip/hip_bf16.h>
#include <math.h>
#include <stdint.h>

typedef _Float16 half_t;
typedef __attribute__((ext_vector_type(16))) _Float16 v16h;
typedef __attribute__((ext_vector_type(8)))  _Float16 v8h;
typedef __attribute__((ext_vector_type(4)))  _Float16 v4h;
typedef __attribute__((ext_vector_type(8)))  float    v8f;

#define HDIM 128
#define THREADS 256

// ---------------------------------------------------------------- utilities
__global__ void zero_f32(float* __restrict__ p, int n) {
    int g = blockIdx.x * blockDim.x + threadIdx.x;
    int stride = gridDim.x * blockDim.x;
    for (int i = g; i < n; i += stride) p[i] = 0.0f;
}

__global__ void deg_count(const int* __restrict__ idx, float* __restrict__ cnt, int nE) {
    int g = blockIdx.x * blockDim.x + threadIdx.x;
    int stride = gridDim.x * blockDim.x;
    for (int i = g; i < nE; i += stride) {
        __hip_atomic_fetch_add(&cnt[idx[i]], 1.0f, __ATOMIC_RELAXED, __HIP_MEMORY_SCOPE_AGENT);
    }
}

__global__ void inv_clamp(float* __restrict__ p, int n) {
    int g = blockIdx.x * blockDim.x + threadIdx.x;
    int stride = gridDim.x * blockDim.x;
    for (int i = g; i < n; i += stride) p[i] = 1.0f / fmaxf(p[i], 1.0f);
}

// Pre-swizzle a K x 128 f32 weight matrix into this file's contiguous-k WMMA B-operand
// layout (f16): B half e of lane: N = ct*16 + (lane&15), K = ks*32 + e + 16*(lane>>4)
__global__ void w_swizzle(const float* __restrict__ W, half_t* __restrict__ swz, int K) {
    int ksteps = K >> 5;
    int total = K * HDIM;
    int g = blockIdx.x * blockDim.x + threadIdx.x;
    int stride = gridDim.x * blockDim.x;
    for (int i = g; i < total; i += stride) {
        int e    = i & 15;
        int lane = (i >> 4) & 31;
        int rest = i >> 9;               // ct*ksteps + ks
        int ks   = rest % ksteps;
        int ct   = rest / ksteps;
        int hi   = lane >> 4;
        int ln   = lane & 15;
        int k    = ks * 32 + e + 16 * hi;
        swz[i] = (half_t)W[k * HDIM + ct * 16 + ln];
    }
}

// ---------------------------------------------------- edge scatter (seg-sum)
// one wave per edge; features are f16 -> 8B gather/lane and packed-f16 atomics
// (global_atomic_pk_add_f16: 2 halves per atomic, half the op count of f32)
__global__ void scatter_accum(const half_t* __restrict__ feat, const int* __restrict__ sidx,
                              const int* __restrict__ didx, half_t* __restrict__ acc, int nE) {
    int g = blockIdx.x * blockDim.x + threadIdx.x;
    int stride = gridDim.x * blockDim.x;
    for (int t = g; t < nE * 32; t += stride) {
        int e = t >> 5;
        int lane = t & 31;
        int s = sidx[e];
        int d = didx[e];
        v4h v = *(const v4h*)&feat[(size_t)s * HDIM + lane * 4];
        uint2 u = __builtin_bit_cast(uint2, v);
        uint64_t a0 = (uint64_t)(uintptr_t)&acc[(size_t)d * HDIM + lane * 4];
        uint64_t a1 = a0 + 4;
        asm volatile("global_atomic_pk_add_f16 %0, %1, off" :: "v"(a0), "v"(u.x) : "memory");
        asm volatile("global_atomic_pk_add_f16 %0, %1, off" :: "v"(a1), "v"(u.y) : "memory");
    }
}

// ------------------------------------------------------------- fused GEMM
// out[N,128](f16) = epi( A1[N,K1]*diag(scale1) @ W1 + (A2[N,128](f16) @ W2) + bias )
// A1 is f32 when A1F16==false (raw inputs), f16 otherwise (aggregates).
// flags: bit0 = row l2-normalize, bit1 = relu
// B held in registers for the whole block; RT row tiles of 16 amortize it.
// A2 staged into LDS with gfx1250 async-LDS loads (ASYNCcnt).
template <int K1, bool HAS_A2, bool A1F16, int RT>
__global__ __launch_bounds__(THREADS) void gemm_fused(
    const void* __restrict__ A1v, const float* __restrict__ scale1,
    const half_t* __restrict__ W1s,
    const half_t* __restrict__ A2, const half_t* __restrict__ W2s,
    const float* __restrict__ bias,
    half_t* __restrict__ out, int N, int flags)
{
    constexpr int KS1 = K1 >> 5;
    static_assert(!A1F16 || K1 == HDIM, "f16 A1 implies K1=128");
    const int tid  = threadIdx.x;
    const int wave = tid >> 5;
    const int lane = tid & 31;
    const int hi   = lane >> 4;
    const int ln   = lane & 15;

    __shared__ half_t sA1[16][HDIM + 8];   // +8 halves pad: conflict-free ds reads
    __shared__ half_t sA2[16][HDIM + 8];
    __shared__ float  sC[16][HDIM + 4];
    __shared__ float  sInv[16];

    // hoist B operands into registers: compile-time offsets -> b128 clauses
    v16h b1[KS1];
    {
        const half_t* bb = &W1s[(((size_t)wave * KS1) * 32 + lane) * 16];
#pragma unroll
        for (int ks = 0; ks < KS1; ++ks) b1[ks] = *(const v16h*)(bb + (size_t)ks * 512);
    }
    v16h b2[4];
    if (HAS_A2) {
        const half_t* bb = &W2s[(((size_t)wave * 4) * 32 + lane) * 16];
#pragma unroll
        for (int ks = 0; ks < 4; ++ks) b2[ks] = *(const v16h*)(bb + (size_t)ks * 512);
    }
    const float bv = bias ? bias[wave * 16 + ln] : 0.0f;

    const int rr = tid >> 4;               // epilogue/stage row 0..15
    const int c8 = (tid & 15) * 8;          // epilogue/stage col chunk (8 halves)

    for (int rt = 0; rt < RT; ++rt) {
        const int row0 = (blockIdx.x * RT + rt) * 16;
        if (row0 >= N) break;  // uniform across block

        // ---- stage A1
        if (A1F16) {
            // one 16B chunk per thread, fold mean scale
            const half_t* A1h = (const half_t*)A1v;
            int gr = row0 + rr;
            v8h h = {};
            if (gr < N) {
                h = *(const v8h*)&A1h[(size_t)gr * HDIM + c8];
                if (scale1) {
                    float s = scale1[gr];
#pragma unroll
                    for (int e = 0; e < 8; ++e) h[e] = (half_t)((float)h[e] * s);
                }
            }
            *(v8h*)&sA1[rr][c8] = h;
        } else {
            const float* A1f = (const float*)A1v;
#pragma unroll
            for (int i = tid; i < 16 * K1; i += THREADS) {
                int r = i / K1, c = i % K1;
                int gr = row0 + r;
                float v = (gr < N) ? A1f[(size_t)gr * K1 + c] : 0.0f;
                sA1[r][c] = (half_t)v;
            }
        }
        // ---- stage A2 via async global->LDS copy (pure f16 byte copy)
        if (HAS_A2) {
            int gr = row0 + rr;
            if (gr < N) {
                uint32_t lds_off = (uint32_t)(uintptr_t)&sA2[rr][c8];
                uint64_t ga = (uint64_t)(uintptr_t)&A2[(size_t)gr * HDIM + c8];
                asm volatile("global_load_async_to_lds_b128 %0, %1, off"
                             :: "v"(lds_off), "v"(ga) : "memory");
            } else {
                v8h z = {};
                *(v8h*)&sA2[rr][c8] = z;
            }
            asm volatile("s_wait_asynccnt 0x0" ::: "memory");
        }
        __syncthreads();

        v8f acc = {};
#pragma unroll
        for (int ks = 0; ks < KS1; ++ks) {
            // A operand: row=ln,
            // halves K = ks*32 + hi*16 + {0..15} (matches B swizzle)
            v8h alo = *(const v8h*)&sA1[ln][ks * 32 + hi * 16];
            v8h ahi = *(const v8h*)&sA1[ln][ks * 32 + hi * 16 + 8];
            v16h a = __builtin_shufflevector(alo, ahi,
                                             0,1,2,3,4,5,6,7,8,9,10,11,12,13,14,15);
            acc = __builtin_amdgcn_wmma_f32_16x16x32_f16(false, a, false, b1[ks],
                                                         (short)0, acc, false, false);
        }
        if (HAS_A2) {
#pragma unroll
            for (int ks = 0; ks < 4; ++ks) {
                v8h alo = *(const v8h*)&sA2[ln][ks * 32 + hi * 16];
                v8h ahi = *(const v8h*)&sA2[ln][ks * 32 + hi * 16 + 8];
                v16h a = __builtin_shufflevector(alo, ahi,
                                                 0,1,2,3,4,5,6,7,8,9,10,11,12,13,14,15);
                acc = __builtin_amdgcn_wmma_f32_16x16x32_f16(false, a, false, b2[ks],
                                                             (short)0, acc, false, false);
            }
        }
#pragma unroll
        for (int r = 0; r < 8; ++r) acc[r] += bv;

        // ---- unified epilogue through LDS (coalesced 16B f16 stores)
        // C/D layout: acc[r] is (row r + 8*hi, col wave*16 + ln)
#pragma unroll
        for (int r = 0; r < 8; ++r) sC[r + 8 * hi][wave * 16 + ln] = acc[r];
        __syncthreads();
        if (tid < 16) {
            float iv = 1.0f;
            if (flags & 1) {
                float s = 0.0f;
                for (int c = 0; c < HDIM; ++c) { float v = sC[tid][c]; s += v * v; }
                iv = 1.0f / fmaxf(sqrtf(s), 1e-12f);
            }
            sInv[tid] = iv;
        }
        __syncthreads();
        {
            int gr = row0 + rr;
            if (gr < N) {
                float iv = sInv[rr];
                v8h o;
#pragma unroll
                for (int e = 0; e < 8; ++e) {
                    float v = sC[rr][c8 + e] * iv;
                    if (flags & 2) v = fmaxf(v, 0.0f);
                    o[e] = (half_t)v;
                }
                *(v8h*)&out[(size_t)gr * HDIM + c8] = o;
            }
        }
        __syncthreads();  // protect sA/sC reuse across row tiles
    }
}

// ----------------------------------------------------------- link predictor
__global__ void dot_pairs(const half_t* __restrict__ zt, const half_t* __restrict__ zp,
                          const int* __restrict__ ls, const int* __restrict__ ld,
                          float* __restrict__ out, int n) {
    int g = blockIdx.x * blockDim.x + threadIdx.x;
    int stride = gridDim.x * blockDim.x;
    for (int t = g; t < n * 32; t += stride) {
        int i = t >> 5, lane = t & 31;
        int s = ls[i], d = ld[i];
        v4h a = *(const v4h*)&zt[(size_t)s * HDIM + lane * 4];
        v4h b = *(const v4h*)&zp[(size_t)d * HDIM + lane * 4];
        float p = 0.0f;
#pragma unroll
        for (int e = 0; e < 4; ++e) p += (float)a[e] * (float)b[e];
#pragma unroll
        for (int off = 16; off; off >>= 1) p += __shfl_xor(p, off, 32);
        if (lane == 0) out[i] = p;
    }
}

// ----------------------------------------------------------------- launcher
extern "C" void kernel_launch(void* const* d_in, const int* in_sizes, int n_in,
                              void* d_out, int out_size, void* d_ws, size_t ws_size,
                              hipStream_t stream) {
    const int FT = 128, FP = 64;
    const int NT = in_sizes[0] / FT;
    const int NP = in_sizes[1] / FP;
    const int E  = in_sizes[2];
    const int EL = in_sizes[4];

    const float* x_t  = (const float*)d_in[0];
    const float* x_p  = (const float*)d_in[1];
    const int* e_src  = (const int*)d_in[2];
    const int* e_dst  = (const int*)d_in[3];
    const int* l_src  = (const int*)d_in[4];
    const int* l_dst  = (const int*)d_in[5];
    const float* Wn_t = (const float*)d_in[6];  const float* bn_t = (const float*)d_in[7];
    const float* Wn_p = (const float*)d_in[8];  const float* bn_p = (const float*)d_in[9];
    const float* W1lp = (const float*)d_in[10]; const float* b1p = (const float*)d_in[11];
    const float* W1rp = (const float*)d_in[12];
    const float* W1lt = (const float*)d_in[13]; const float* b1t = (const float*)d_in[14];
    const float* W1rt = (const float*)d_in[15];
    const float* W2lp = (const float*)d_in[16]; const float* b2p = (const float*)d_in[17];
    const float* W2rp = (const float*)d_in[18];
    const float* W2lt = (const float*)d_in[19]; const float* b2t = (const float*)d_in[20];
    const float* W2rt = (const float*)d_in[21];

    // workspace carve-out (256B aligned)
    char* base = (char*)d_ws;
    size_t off = 0;
    auto carve = [&](size_t bytes) -> void* {
        void* p = base + off;
        off = (off + bytes + 255) & ~(size_t)255;
        return p;
    };
    half_t* sWn_t = (half_t*)carve((size_t)128 * 128 * 2);
    half_t* sWn_p = (half_t*)carve((size_t)64  * 128 * 2);
    half_t* sW1lp = (half_t*)carve((size_t)128 * 128 * 2);
    half_t* sW1rp = (half_t*)carve((size_t)128 * 128 * 2);
    half_t* sW1lt = (half_t*)carve((size_t)128 * 128 * 2);
    half_t* sW1rt = (half_t*)carve((size_t)128 * 128 * 2);
    half_t* sW2lp = (half_t*)carve((size_t)128 * 128 * 2);
    half_t* sW2rp = (half_t*)carve((size_t)128 * 128 * 2);
    half_t* sW2lt = (half_t*)carve((size_t)128 * 128 * 2);
    half_t* sW2rt = (half_t*)carve((size_t)128 * 128 * 2);
    float* inv_p = (float*)carve((size_t)NP * 4);
    float* inv_t = (float*)carve((size_t)NT * 4);
    // f16 feature buffers
    half_t* hb_t = (half_t*)carve((size_t)NT * HDIM * 2);   // h_t, later z_t
    half_t* ab_t = (half_t*)carve((size_t)NT * HDIM * 2);   // agg_t (both layers)
    half_t* ob_t = (half_t*)carve((size_t)NT * HDIM * 2);   // o_t
    half_t* hb_p = (half_t*)carve((size_t)NP * HDIM * 2);   // h_p, later z_p
    half_t* ab_p = (half_t*)carve((size_t)NP * HDIM * 2);   // agg_p (both layers)
    half_t* ob_p = (half_t*)carve((size_t)NP * HDIM * 2);   // o_p
    (void)ws_size; (void)n_in; (void)out_size;

    const int ZB = 2048, SB = 8192, WB = 64;
    constexpr int RT = 4;
    const int gT = (NT + 16 * RT - 1) / (16 * RT);
    const int gP = (NP + 16 * RT - 1) / (16 * RT);

    // degrees -> inverse counts
    zero_f32<<<ZB, THREADS, 0, stream>>>(inv_p, NP);
    zero_f32<<<ZB, THREADS, 0, stream>>>(inv_t, NT);
    deg_count<<<SB, THREADS, 0, stream>>>(e_dst, inv_p, E);
    deg_count<<<SB, THREADS, 0, stream>>>(e_src, inv_t, E);
    inv_clamp<<<ZB, THREADS, 0, stream>>>(inv_p, NP);
    inv_clamp<<<ZB, THREADS, 0, stream>>>(inv_t, NT);

    // swizzle weights to f16 WMMA B layout
    w_swizzle<<<WB, THREADS, 0, stream>>>(Wn_t, sWn_t, 128);
    w_swizzle<<<WB, THREADS, 0, stream>>>(Wn_p, sWn_p, 64);
    w_swizzle<<<WB, THREADS, 0, stream>>>(W1lp, sW1lp, 128);
    w_swizzle<<<WB, THREADS, 0, stream>>>(W1rp, sW1rp, 128);
    w_swizzle<<<WB, THREADS, 0, stream>>>(W1lt, sW1lt, 128);
    w_swizzle<<<WB, THREADS, 0, stream>>>(W1rt, sW1rt, 128);
    w_swizzle<<<WB, THREADS, 0, stream>>>(W2lp, sW2lp, 128);
    w_swizzle<<<WB, THREADS, 0, stream>>>(W2rp, sW2rp, 128);
    w_swizzle<<<WB, THREADS, 0, stream>>>(W2lt, sW2lt, 128);
    w_swizzle<<<WB, THREADS, 0, stream>>>(W2rt, sW2rt, 128);

    // input projections (f32 inputs -> f16 features)
    gemm_fused<128, false, false, RT><<<gT, THREADS, 0, stream>>>(
        x_t, nullptr, sWn_t, nullptr, nullptr, bn_t, hb_t, NT, 0);
    gemm_fused<64, false, false, RT><<<gP, THREADS, 0, stream>>>(
        x_p, nullptr, sWn_p, nullptr, nullptr, bn_p, hb_p, NP, 0);

    // layer 1: aggregate (f16 pk atomics) then fused dual-GEMM + l2norm + relu
    zero_f32<<<ZB, THREADS, 0, stream>>>((float*)ab_p, NP * (HDIM / 2));
    zero_f32<<<ZB, THREADS, 0, stream>>>((float*)ab_t, NT * (HDIM / 2));
    scatter_accum<<<SB, THREADS, 0, stream>>>(hb_t, e_src, e_dst, ab_p, E);
    scatter_accum<<<SB, THREADS, 0, stream>>>(hb_p, e_dst, e_src, ab_t, E);
    gemm_fused<128, true, true, RT><<<gP, THREADS, 0, stream>>>(
        ab_p, inv_p, sW1lp, hb_p, sW1rp, b1p, ob_p, NP, 3);
    gemm_fused<128, true, true, RT><<<gT, THREADS, 0, stream>>>(
        ab_t, inv_t, sW1lt, hb_t, sW1rt, b1t, ob_t, NT, 3);

    // layer 2: re-aggregate, l2norm (no relu); z_* overwrite h_* buffers
    zero_f32<<<ZB, THREADS, 0, stream>>>((float*)ab_p, NP * (HDIM / 2));
    zero_f32<<<ZB, THREADS, 0, stream>>>((float*)ab_t, NT * (HDIM / 2));
    scatter_accum<<<SB, THREADS, 0, stream>>>(ob_t, e_src, e_dst, ab_p, E);
    scatter_accum<<<SB, THREADS, 0, stream>>>(ob_p, e_dst, e_src, ab_t, E);
    gemm_fused<128, true, true, RT><<<gP, THREADS, 0, stream>>>(
        ab_p, inv_p, sW2lp, ob_p, sW2rp, b2p, hb_p, NP, 1);
    gemm_fused<128, true, true, RT><<<gT, THREADS, 0, stream>>>(
        ab_t, inv_t, sW2lt, ob_t, sW2rt, b2t, hb_t, NT, 1);

    // link predictor
    dot_pairs<<<SB, THREADS, 0, stream>>>(hb_t, hb_p, l_src, l_dst, (float*)d_out, EL);
}
